// MultiHeadAttention_37933151158411
// MI455X (gfx1250) — matrix operand order repaired and checked by reading, not compiled, not run on hardware
//
#include <hip/hip_runtime.h>
#include <hip/hip_bf16.h>

typedef __bf16 bf16_t;
typedef __attribute__((ext_vector_type(16))) __bf16 v16bf;
typedef __attribute__((ext_vector_type(8)))  __bf16 v8bf;
typedef __attribute__((ext_vector_type(4)))  __bf16 v4bf;
typedef __attribute__((ext_vector_type(8)))  float  v8f;

#define B_     4
#define L_     1024
#define D_     1024
#define H_     16
#define HD_    64
#define INNER_ 1024
#define QKV_LD 3072
#define LN_EPS 1e-5f

static __device__ __forceinline__ v8f wmma_bf16(v16bf a, v16bf b, v8f c) {
  // D(f32 16x16) = A(bf16 16x32) * B(bf16 32x16) + C
  return __builtin_amdgcn_wmma_f32_16x16x32_bf16(false, a, false, b, (short)0, c,
                                                 false, false);
}

static __device__ __forceinline__ v16bf cat8(v8bf lo, v8bf hi) {
  return __builtin_shufflevector(lo, hi, 0,1,2,3,4,5,6,7,8,9,10,11,12,13,14,15);
}

// ---------------------------------------------------------------------------
// f32 -> bf16 conversion (4 elements per thread)
// ---------------------------------------------------------------------------
__global__ void cvt_f32_bf16(const float* __restrict__ s, bf16_t* __restrict__ d, int n) {
  int i = 4 * (blockIdx.x * blockDim.x + threadIdx.x);
  if (i + 3 < n) {
    float4 v = *(const float4*)(s + i);
    v4bf o;
    o[0] = (bf16_t)v.x; o[1] = (bf16_t)v.y; o[2] = (bf16_t)v.z; o[3] = (bf16_t)v.w;
    *(v4bf*)(d + i) = o;
  }
}

// ---------------------------------------------------------------------------
// WMMA GEMM: C[M,N] = A[M,K](row-major bf16) * W[N,K](row-major bf16)^T
// torch Linear layout [out,in] => W rows are contraction-contiguous: B-frags
// are plain contiguous 16-element loads (lane = N column, lane-half = K half).
// Block: 256 threads = 8 waves; wave w -> rows [by*128 + 16w, +16), cols [bx*64, +64)
// ---------------------------------------------------------------------------
template<bool OUT_BF16>
__global__ void gemm_wmma(const bf16_t* __restrict__ A, const bf16_t* __restrict__ W,
                          float* __restrict__ Cf, bf16_t* __restrict__ Cb,
                          const float* __restrict__ bias, int M, int N, int K) {
  const int lane = threadIdx.x & 31;
  const int wave = threadIdx.x >> 5;
  const int n    = lane & 15;     // A-frag row index / B-frag column / C column
  const int h    = lane >> 4;     // lane half
  const int rowm = blockIdx.y * 128 + wave * 16 + n;   // A-frag row (m = lane&15)
  const int col0 = blockIdx.x * 64;

  const bf16_t* ar = A + (size_t)rowm * K;
  v8f acc[4] = {};

  for (int k0 = 0; k0 < K; k0 += 32) {
    __builtin_prefetch(ar + k0 + 256, 0, 1);
    // A frag: lane-half h holds K = {k0+16h..+16} (contiguous)
    v8bf alo = *(const v8bf*)(ar + k0 + 16 * h);
    v8bf ahi = *(const v8bf*)(ar + k0 + 16 * h + 8);
    v16bf af = cat8(alo, ahi);
#pragma unroll
    for (int t = 0; t < 4; ++t) {
      // B frag: column (col0+16t+n), K = k0+16h+e (contiguous in W row)
      const bf16_t* wr = W + (size_t)(col0 + 16 * t + n) * K + k0 + 16 * h;
      v16bf wv = *(const v16bf*)wr;
      acc[t] = wmma_bf16(af, wv, acc[t]);
    }
  }

#pragma unroll
  for (int t = 0; t < 4; ++t) {
    int col = col0 + 16 * t + n;
#pragma unroll
    for (int r = 0; r < 8; ++r) {
      int rr = blockIdx.y * 128 + wave * 16 + 8 * h + r;  // C rows are M=8h+r
      size_t idx = (size_t)rr * N + col;
      if (OUT_BF16) Cb[idx] = (bf16_t)acc[t][r];
      else          Cf[idx] = acc[t][r] + bias[col];
    }
  }
}

// ---------------------------------------------------------------------------
// Flash attention: one block per (b, head, 128-row query tile); 8 waves,
// each wave owns 16 query rows. Key/value tiles of 64 columns.
// qkv layout: [B, L, 3*INNER] bf16 (Q at 0, K at INNER, V at 2*INNER).
// ---------------------------------------------------------------------------
__global__ void attn_flash(const bf16_t* __restrict__ qkv,
                           const unsigned char* __restrict__ maskp,
                           bf16_t* __restrict__ attn) {
  __shared__ bf16_t Vt[64 * 64];       // V tile transposed: Vt[d][j]
  __shared__ bf16_t P[8 * 16 * 64];    // per-wave P tiles, row-major [i][j]

  const int lane = threadIdx.x & 31;
  const int wave = threadIdx.x >> 5;
  const int n = lane & 15, h = lane >> 4;
  const int b = blockIdx.z, hd = blockIdx.y, it = blockIdx.x;
  const int i0 = it * 128;
  const size_t base = ((size_t)b * L_) * QKV_LD + (size_t)hd * HD_;

  // Persistent Q A-fragments (HD=64 -> two K=32 steps)
  const bf16_t* qr = qkv + base + (size_t)(i0 + wave * 16 + n) * QKV_LD;
  v16bf qf[2];
#pragma unroll
  for (int kk = 0; kk < 2; ++kk) {
    v8bf lo = *(const v8bf*)(qr + 32 * kk + 16 * h);
    v8bf hi = *(const v8bf*)(qr + 32 * kk + 16 * h + 8);
    qf[kk] = cat8(lo, hi);
  }

  v8f O[4] = {};
  float mrow[8], lrow[8];
#pragma unroll
  for (int r = 0; r < 8; ++r) { mrow[r] = -1e30f; lrow[r] = 0.f; }

  bf16_t* pw = P + wave * (16 * 64);

  for (int jt = 0; jt < L_ / 64; ++jt) {
    const int j0 = jt * 64;

    __syncthreads();
    {   // cooperative transposed staging of V tile: Vt[d][j] = V[j0+j][d]
      int t = threadIdx.x;
      int j = t >> 2;
      int d0 = (t & 3) * 16;
      const bf16_t* vr = qkv + base + (size_t)(j0 + j) * QKV_LD + 2 * INNER_ + d0;
      v16bf vv = *(const v16bf*)vr;
#pragma unroll
      for (int e = 0; e < 16; ++e) Vt[(d0 + e) * 64 + j] = vv[e];
    }
    __syncthreads();

    // S = Q * K^T  (K rows are contraction-contiguous -> direct B-frag loads)
    v8f S[4] = {};
#pragma unroll
    for (int nt = 0; nt < 4; ++nt) {
      const bf16_t* kr = qkv + base + (size_t)(j0 + 16 * nt + n) * QKV_LD + INNER_;
#pragma unroll
      for (int kk = 0; kk < 2; ++kk) {
        v16bf kf = *(const v16bf*)(kr + 32 * kk + 16 * h);
        S[nt] = wmma_bf16(qf[kk], kf, S[nt]);
      }
    }

    // scale + mask (mask per key column)
#pragma unroll
    for (int nt = 0; nt < 4; ++nt) {
      unsigned char mv = maskp[(size_t)b * L_ + j0 + 16 * nt + n];
#pragma unroll
      for (int r = 0; r < 8; ++r) {
        float s = S[nt][r] * 0.125f;           // 1/sqrt(64)
        S[nt][r] = mv ? s : -1e30f;
      }
    }

    // online softmax: rows live at M=8h+r; columns across lanes n=0..15
    float alpha[8];
#pragma unroll
    for (int r = 0; r < 8; ++r) {
      float mx = fmaxf(fmaxf(S[0][r], S[1][r]), fmaxf(S[2][r], S[3][r]));
      mx = fmaxf(mx, __shfl_xor(mx, 1));
      mx = fmaxf(mx, __shfl_xor(mx, 2));
      mx = fmaxf(mx, __shfl_xor(mx, 4));
      mx = fmaxf(mx, __shfl_xor(mx, 8));       // bit4 (lane half) untouched
      float mnew = fmaxf(mrow[r], mx);
      alpha[r] = __expf(mrow[r] - mnew);
      mrow[r] = mnew;
    }
#pragma unroll
    for (int r = 0; r < 8; ++r) {
      float sum = 0.f;
#pragma unroll
      for (int nt = 0; nt < 4; ++nt) {
        float p = __expf(S[nt][r] - mrow[r]);
        S[nt][r] = p;
        sum += p;
      }
      sum += __shfl_xor(sum, 1); sum += __shfl_xor(sum, 2);
      sum += __shfl_xor(sum, 4); sum += __shfl_xor(sum, 8);
      lrow[r] = lrow[r] * alpha[r] + sum;
#pragma unroll
      for (int nt = 0; nt < 4; ++nt) O[nt][r] *= alpha[r];
    }

    // C-layout -> A-layout reshape of P through wave-private LDS
#pragma unroll
    for (int nt = 0; nt < 4; ++nt)
#pragma unroll
      for (int r = 0; r < 8; ++r)
        pw[(8 * h + r) * 64 + 16 * nt + n] = (bf16_t)S[nt][r];
    asm volatile("s_wait_dscnt 0x0" ::: "memory");   // wave-local, no barrier needed

    v16bf pf[2];
#pragma unroll
    for (int kk = 0; kk < 2; ++kk) {
      v8bf lo = *(const v8bf*)(pw + n * 64 + 32 * kk + 16 * h);
      v8bf hi = *(const v8bf*)(pw + n * 64 + 32 * kk + 16 * h + 8);
      pf[kk] = cat8(lo, hi);
    }

    // O += P * V  (B-frags = contiguous rows of transposed V tile in LDS)
#pragma unroll
    for (int nt = 0; nt < 4; ++nt) {
#pragma unroll
      for (int kk = 0; kk < 2; ++kk) {
        v16bf vf = *(const v16bf*)&Vt[(16 * nt + n) * 64 + 32 * kk + 16 * h];
        O[nt] = wmma_bf16(pf[kk], vf, O[nt]);
      }
    }
  }

  // normalize and emit [B, L, H*HD] bf16 for the output projection
#pragma unroll
  for (int r = 0; r < 8; ++r) {
    float inv = 1.0f / lrow[r];
    int row = i0 + wave * 16 + 8 * h + r;
#pragma unroll
    for (int nt = 0; nt < 4; ++nt)
      attn[((size_t)b * L_ + row) * INNER_ + hd * HD_ + 16 * nt + n] =
          (bf16_t)(O[nt][r] * inv);
  }
}

// ---------------------------------------------------------------------------
// In-place LayerNorm over last dim (D=1024), one block per row
// ---------------------------------------------------------------------------
__global__ void layernorm_inplace(float* __restrict__ y, const float* __restrict__ gamma,
                                  const float* __restrict__ beta) {
  __shared__ float red[16];
  float* p = y + (size_t)blockIdx.x * D_;
  const int t = threadIdx.x;
  float v[4], s = 0.f, s2 = 0.f;
#pragma unroll
  for (int i = 0; i < 4; ++i) {
    v[i] = p[t + 256 * i];
    s += v[i]; s2 += v[i] * v[i];
  }
  for (int off = 1; off < 32; off <<= 1) {
    s  += __shfl_xor(s,  off);
    s2 += __shfl_xor(s2, off);
  }
  if ((t & 31) == 0) { red[t >> 5] = s; red[8 + (t >> 5)] = s2; }
  __syncthreads();
  if (t == 0) {
    float S = 0.f, S2 = 0.f;
    for (int w = 0; w < 8; ++w) { S += red[w]; S2 += red[8 + w]; }
    red[0] = S; red[8] = S2;
  }
  __syncthreads();
  float mu   = red[0] / (float)D_;
  float var  = red[8] / (float)D_ - mu * mu;
  float rstd = rsqrtf(var + LN_EPS);
#pragma unroll
  for (int i = 0; i < 4; ++i) {
    int d = t + 256 * i;
    p[d] = (v[i] - mu) * rstd * gamma[d] + beta[d];
  }
}

// ---------------------------------------------------------------------------
extern "C" void kernel_launch(void* const* d_in, const int* in_sizes, int n_in,
                              void* d_out, int out_size, void* d_ws, size_t ws_size,
                              hipStream_t stream) {
  const float* x            = (const float*)d_in[0];
  const unsigned char* mask = (const unsigned char*)d_in[1];
  const float* w_qkv        = (const float*)d_in[2];
  const float* w_out        = (const float*)d_in[3];
  const float* b_out        = (const float*)d_in[4];
  const float* gamma        = (const float*)d_in[5];
  const float* beta         = (const float*)d_in[6];
  float* out = (float*)d_out;

  char* ws = (char*)d_ws;
  bf16_t* xb    = (bf16_t*)(ws);                          //  8 MB: x bf16
  bf16_t* wqkvb = (bf16_t*)(ws + (size_t)(8)  * 1048576); //  6 MB: w_qkv bf16
  bf16_t* woutb = (bf16_t*)(ws + (size_t)(14) * 1048576); //  2 MB: w_out bf16
  bf16_t* qkvb  = (bf16_t*)(ws + (size_t)(16) * 1048576); // 24 MB: qkv bf16
  bf16_t* attnb = (bf16_t*)(ws + (size_t)(40) * 1048576); //  8 MB: attn out bf16

  const int nx = B_ * L_ * D_;        // 4194304
  const int nq = 3 * INNER_ * D_;     // 3145728
  const int no = D_ * INNER_;         // 1048576
  cvt_f32_bf16<<<nx / 1024, 256, 0, stream>>>(x,     xb,    nx);
  cvt_f32_bf16<<<nq / 1024, 256, 0, stream>>>(w_qkv, wqkvb, nq);
  cvt_f32_bf16<<<no / 1024, 256, 0, stream>>>(w_out, woutb, no);

  // qkv = x @ w_qkv^T   [4096 x 3072]
  gemm_wmma<true><<<dim3(QKV_LD / 64, (B_ * L_) / 128), 256, 0, stream>>>(
      xb, wqkvb, nullptr, qkvb, nullptr, B_ * L_, QKV_LD, D_);

  // fused flash attention
  attn_flash<<<dim3(L_ / 128, H_, B_), 256, 0, stream>>>(qkvb, mask, attnb);

  // out = attn @ w_out^T + b_out   [4096 x 1024]
  gemm_wmma<false><<<dim3(D_ / 64, (B_ * L_) / 128), 256, 0, stream>>>(
      attnb, woutb, out, nullptr, b_out, B_ * L_, D_, INNER_);

  layernorm_inplace<<<B_ * L_, 256, 0, stream>>>(out, gamma, beta);
}
